// NLBasicBlock_14714557956618
// MI455X (gfx1250) — matrix operand order repaired and checked by reading, not compiled, not run on hardware
//
#include <hip/hip_runtime.h>

typedef __attribute__((ext_vector_type(16))) __bf16 v16bf;
typedef __attribute__((ext_vector_type(8)))  float  v8f;

#define BATCH 2
#define CH    64
#define HH    80
#define WW    80
#define NPIX  6400          // 80*80
#define NT16  400           // NPIX/16
#define NT32  200           // NPIX/32

// ---------- helpers ----------------------------------------------------------

static __device__ inline unsigned short f2bf(float f) {
    union { float f; unsigned u; } v; v.f = f;
    unsigned u = v.u;
    unsigned r = (u + 0x7FFFu + ((u >> 16) & 1u)) >> 16;   // RNE
    return (unsigned short)r;
}
static __device__ inline __bf16 u2bf(unsigned short u) {
    union { unsigned short u; __bf16 b; } x; x.u = u; return x.b;
}
static __device__ inline v8f wmma_bf16(v16bf a, v16bf b, v8f c) {
    return __builtin_amdgcn_wmma_f32_16x16x32_bf16(false, a, false, b, (short)0, c, false, false);
}
static __device__ inline int reflect80(int v) {
    if (v < 0)   return -v;          // -1 -> 1
    if (v >= 80) return 158 - v;     // 80 -> 78
    return v;
}
// A-fragment (16x32 bf16) element index for K value kk (0..31)
static __device__ inline void a_he(int kk, int& h, int& e) {
    if (kk < 16) { h = kk >> 3; e = kk & 7; }
    else         { int t = kk - 16; h = t >> 3; e = 8 + (t & 7); }
}

// gfx1250 async global->LDS copy (ASYNCcnt-tracked DMA, no VGPR round trip).
// lds_addr: 32-bit LDS byte address (low 32 bits of the generic pointer);
// gsrc: 64-bit global address.  16 bytes per lane.
static __device__ inline void async_copy_b128(unsigned lds_addr, const void* gsrc) {
    asm volatile("global_load_async_to_lds_b128 %0, %1, off"
                 :: "v"(lds_addr), "v"(gsrc) : "memory");
}
static __device__ inline void wait_async0() {
    asm volatile("s_wait_asynccnt 0" ::: "memory");
}
static __device__ inline unsigned lds_off(const void* p) {
    return (unsigned)(unsigned long long)p;   // LDS aperture keeps offset in addr[31:0]
}

// ---------- K1/K6: 3x3 conv, reflect pad, PReLU, optional raw output ---------------------
__global__ __launch_bounds__(256)
void conv3x3_kernel(const float* __restrict__ in, const float* __restrict__ w,
                    const float* __restrict__ alpha_p,
                    float* __restrict__ out_raw, float* __restrict__ out_act) {
    __shared__ float sw[16 * 576];
    __shared__ float sin[16 * 18 * 18];

    const int tx = threadIdx.x, ty = threadIdx.y;
    const int tid = ty * 16 + tx;
    const int b  = blockIdx.z >> 2;
    const int ob = (blockIdx.z & 3) * 16;
    const int h0 = blockIdx.y * 16, w0 = blockIdx.x * 16;

    for (int idx = tid; idx < 16 * 576; idx += 256) {
        int o = idx / 576, rest = idx % 576;
        sw[idx] = w[(ob + o) * 576 + rest];
    }

    float acc[16];
#pragma unroll
    for (int o = 0; o < 16; ++o) acc[o] = 0.f;

    for (int cc = 0; cc < 4; ++cc) {
        __syncthreads();
        for (int idx = tid; idx < 16 * 324; idx += 256) {
            int ci = idx / 324, r = idx % 324;
            int py = r / 18, px = r % 18;
            int gh = reflect80(h0 - 1 + py);
            int gw = reflect80(w0 - 1 + px);
            sin[idx] = in[((b * CH + cc * 16 + ci) * HH + gh) * WW + gw];
        }
        __syncthreads();
        for (int ci = 0; ci < 16; ++ci) {
#pragma unroll
            for (int tap = 0; tap < 9; ++tap) {
                int dy = tap / 3, dx = tap % 3;
                float xv = sin[ci * 324 + (ty + dy) * 18 + (tx + dx)];
                int wb = (cc * 16 + ci) * 9 + tap;
#pragma unroll
                for (int o = 0; o < 16; ++o)
                    acc[o] = fmaf(xv, sw[o * 576 + wb], acc[o]);
            }
        }
    }

    const float alpha = *alpha_p;
    const int px = (h0 + ty) * WW + (w0 + tx);
#pragma unroll
    for (int o = 0; o < 16; ++o) {
        float v = acc[o];
        long off = (long)(b * CH + ob + o) * NPIX + px;
        if (out_raw) out_raw[off] = v;
        out_act[off] = (v >= 0.f) ? v : alpha * v;
    }
}

// ---------- K2: 1x1 projections theta/phi/g, packed into WMMA fragments -----
__global__ __launch_bounds__(256)
void proj_pack_kernel(const float* __restrict__ p,
                      const float* __restrict__ wg, const float* __restrict__ wth,
                      const float* __restrict__ wph,
                      unsigned short* __restrict__ thT,   // A-frag pack [b][it][chunk][lane][16]
                      unsigned short* __restrict__ phiP,  // B-frag pack [b][jt][chunk][lane][16]
                      unsigned short* __restrict__ gTP) { // B-frag pack [b][jc32][ct][lane][16]
    __shared__ float swg[4096], swth[4096], swph[4096];
    const int tid = threadIdx.x;
    for (int i = tid; i < 4096; i += 256) { swg[i] = wg[i]; swth[i] = wth[i]; swph[i] = wph[i]; }
    __syncthreads();

    const int gid = blockIdx.x * 256 + tid;
    const int b = gid / NPIX, n = gid % NPIX;
    float acc[64];

    // ---- theta ----
#pragma unroll
    for (int o = 0; o < 64; ++o) acc[o] = 0.f;
    for (int c = 0; c < 64; ++c) {
        float x = p[(b * CH + c) * NPIX + n];
#pragma unroll
        for (int o = 0; o < 64; ++o) acc[o] = fmaf(swth[o * 64 + c], x, acc[o]);
    }
    {
        int tile = n >> 4, m = n & 15;
#pragma unroll
        for (int c = 0; c < 64; ++c) {
            int chunk = c >> 5, kk = c & 31, h, e; a_he(kk, h, e);
            int lane = m + 16 * h;
            thT[(((b * NT16 + tile) * 2 + chunk) * 32 + lane) * 16 + e] = f2bf(acc[c]);
        }
    }
    // ---- phi ----
#pragma unroll
    for (int o = 0; o < 64; ++o) acc[o] = 0.f;
    for (int c = 0; c < 64; ++c) {
        float x = p[(b * CH + c) * NPIX + n];
#pragma unroll
        for (int o = 0; o < 64; ++o) acc[o] = fmaf(swph[o * 64 + c], x, acc[o]);
    }
    {
        int tile = n >> 4, jn = n & 15;
#pragma unroll
        for (int c = 0; c < 64; ++c) {
            int chunk = c >> 5, kk = c & 31;
            int h, e; a_he(kk, h, e);
            int lane = jn + 16 * h;
            phiP[(((b * NT16 + tile) * 2 + chunk) * 32 + lane) * 16 + e] = f2bf(acc[c]);
        }
    }
    // ---- g ----
#pragma unroll
    for (int o = 0; o < 64; ++o) acc[o] = 0.f;
    for (int c = 0; c < 64; ++c) {
        float x = p[(b * CH + c) * NPIX + n];
#pragma unroll
        for (int o = 0; o < 64; ++o) acc[o] = fmaf(swg[o * 64 + c], x, acc[o]);
    }
    {
        int jc32 = n >> 5, kk = n & 31;
        int h, e; a_he(kk, h, e);
#pragma unroll
        for (int c = 0; c < 64; ++c) {
            int ct = c >> 4, cn = c & 15;
            int lane = cn + 16 * h;
            gTP[(((b * NT32 + jc32) * 4 + ct) * 32 + lane) * 16 + e] = f2bf(acc[c]);
        }
    }
}

// ---------- K3: column softmax stats -----------------------------------------
// One wave per 16-col tile. theta^T A-fragments are identical across the 8
// waves of a block -> stage them once per block via async DMA, double-buffered.
__global__ __launch_bounds__(256)
void colstats_kernel(const unsigned short* __restrict__ thT,
                     const unsigned short* __restrict__ phiP,
                     float* __restrict__ colm, float* __restrict__ colr) {
    __shared__ __align__(16) unsigned short stageA[2][2048];  // 2 x 4KB: 2 i-tiles x 2 chunks
    __shared__ float sm[8][32], ss[8][32];
    const int tid = threadIdx.x;
    const int lane = tid & 31, wid = tid >> 5;
    const int wg = blockIdx.x * 8 + wid;          // 0..799
    const int b = wg / NT16, jt = wg % NT16;      // b uniform per block

    const v16bf* phiF = (const v16bf*)phiP;
    v16bf b0 = phiF[((b * NT16 + jt) * 2 + 0) * 32 + lane];
    v16bf b1 = phiF[((b * NT16 + jt) * 2 + 1) * 32 + lane];

    const char* Abase = (const char*)thT + (size_t)(b * NT16) * 2048;

    // prefetch first pair of i-tiles
    async_copy_b128(lds_off(&stageA[0][0]) + tid * 16, Abase + tid * 16);

    float m = -INFINITY, s = 0.f;
    for (int it = 0; it < NT16; it += 2) {
        const int buf = (it >> 1) & 1;
        wait_async0();
        __syncthreads();                                  // staged tiles valid
        if (it + 2 < NT16)
            async_copy_b128(lds_off(&stageA[buf ^ 1][0]) + tid * 16,
                            Abase + (size_t)(it + 2) * 1024 + tid * 16);
        const v16bf* sA = (const v16bf*)&stageA[buf][0];
#pragma unroll
        for (int t = 0; t < 2; ++t) {
            v16bf a0 = sA[t * 64 + lane];
            v16bf a1 = sA[t * 64 + 32 + lane];
            v8f f = {};
            f = wmma_bf16(a0, b0, f);
            f = wmma_bf16(a1, b1, f);
            float cm = f[0];
#pragma unroll
            for (int r = 1; r < 8; ++r) cm = fmaxf(cm, f[r]);
            float mn = fmaxf(m, cm);
            float ls = 0.f;
#pragma unroll
            for (int r = 0; r < 8; ++r) ls += __expf(f[r] - mn);
            s = s * __expf(m - mn) + ls;
            m = mn;
        }
    }
    sm[wid][lane] = m; ss[wid][lane] = s;
    __syncthreads();
    if (lane < 16) {
        float ma = sm[wid][lane], mb = sm[wid][lane + 16];
        float M = fmaxf(ma, mb);
        float S = ss[wid][lane] * __expf(ma - M) + ss[wid][lane + 16] * __expf(mb - M);
        colm[b * NPIX + jt * 16 + lane] = M;
        colr[b * NPIX + jt * 16 + lane] = 1.0f / S;
    }
}

// ---------- K4: streaming attention  y = softmax_col(theta^T phi) * g^T ------
// One wave per 16-row tile. phi/gT B-fragments are identical across the 8
// waves of a block -> async-stage 8KB per 32-col chunk, double-buffered.
__global__ __launch_bounds__(256)
void attn_kernel(const unsigned short* __restrict__ thT,
                 const unsigned short* __restrict__ phiP,
                 const unsigned short* __restrict__ gTP,
                 const float* __restrict__ colm, const float* __restrict__ colr,
                 float* __restrict__ y) {
    __shared__ __align__(16) unsigned short stageP[2][2048];  // phi: 2 j-tiles x 2 chunks
    __shared__ __align__(16) unsigned short stageG[2][2048];  // gT: 4 c-tiles
    __shared__ unsigned short Pt[8][16 * 34];                 // per-wave P relayout
    const int tid = threadIdx.x;
    const int lane = tid & 31, wid = tid >> 5;
    const int h = lane >> 4, ln = lane & 15;
    const int wg = blockIdx.x * 8 + wid;
    const int b = wg / NT16, itile = wg % NT16;               // b uniform per block

    const v16bf* thF = (const v16bf*)thT;
    unsigned short* pt = &Pt[wid][0];

    v16bf aF0 = thF[((b * NT16 + itile) * 2 + 0) * 32 + lane];
    v16bf aF1 = thF[((b * NT16 + itile) * 2 + 1) * 32 + lane];

    const char* Pbase = (const char*)phiP + (size_t)(b * NT16) * 2048;
    const char* Gbase = (const char*)gTP + (size_t)(b * NT32) * 4096;

    // prefetch chunk 0
    async_copy_b128(lds_off(&stageP[0][0]) + tid * 16, Pbase + tid * 16);
    async_copy_b128(lds_off(&stageG[0][0]) + tid * 16, Gbase + tid * 16);

    v8f ya[4] = {{}, {}, {}, {}};

    for (int jc = 0; jc < NT32; ++jc) {
        const int buf = jc & 1;
        wait_async0();
        __syncthreads();                                  // staged chunk valid
        if (jc + 1 < NT32) {
            async_copy_b128(lds_off(&stageP[buf ^ 1][0]) + tid * 16,
                            Pbase + (size_t)(jc + 1) * 4096 + tid * 16);
            async_copy_b128(lds_off(&stageG[buf ^ 1][0]) + tid * 16,
                            Gbase + (size_t)(jc + 1) * 4096 + tid * 16);
        }
        const v16bf* sP = (const v16bf*)&stageP[buf][0];
        const v16bf* sG = (const v16bf*)&stageG[buf][0];

        // f tiles (16 rows x 32 cols) via 4 WMMAs
        v8f f0 = {}, f1 = {};
        f0 = wmma_bf16(aF0, sP[lane], f0);
        f0 = wmma_bf16(aF1, sP[32 + lane], f0);
        f1 = wmma_bf16(aF0, sP[64 + lane], f1);
        f1 = wmma_bf16(aF1, sP[96 + lane], f1);

        const int jt0 = jc * 2, jt1 = jc * 2 + 1;
        float m0 = colm[b * NPIX + jt0 * 16 + ln];
        float r0 = colr[b * NPIX + jt0 * 16 + ln];
        float m1 = colm[b * NPIX + jt1 * 16 + ln];
        float r1 = colr[b * NPIX + jt1 * 16 + ln];

        __syncthreads();                       // prior iter's Pt reads done
#pragma unroll
        for (int r = 0; r < 8; ++r) {
            int M = h * 8 + r;
            pt[M * 34 + ln]      = f2bf(__expf(f0[r] - m0) * r0);
            pt[M * 34 + 16 + ln] = f2bf(__expf(f1[r] - m1) * r1);
        }
        __syncthreads();
        // repack P (C-layout) -> A-fragment layout
        v16bf pa;
#pragma unroll
        for (int v = 0; v < 8; ++v) {
            int kbase = (v < 4) ? (8 * h + 2 * v) : (16 + 8 * h + 2 * (v - 4));
            pa[2 * v]     = u2bf(pt[ln * 34 + kbase]);
            pa[2 * v + 1] = u2bf(pt[ln * 34 + kbase + 1]);
        }
        // y += P * gT : 4 WMMAs (64 output channels)
#pragma unroll
        for (int ct = 0; ct < 4; ++ct)
            ya[ct] = wmma_bf16(pa, sG[ct * 32 + lane], ya[ct]);
    }

#pragma unroll
    for (int ct = 0; ct < 4; ++ct)
#pragma unroll
        for (int r = 0; r < 8; ++r) {
            int i = itile * 16 + 8 * h + r;
            int c = ct * 16 + ln;
            y[((long)b * NPIX + i) * 64 + c] = ya[ct][r];
        }
}

// ---------- K5: z = wW*y + p + x1 --------------------------------------------
__global__ __launch_bounds__(256)
void wy_residual_kernel(const float* __restrict__ y, const float* __restrict__ wW,
                        const float* __restrict__ p, const float* __restrict__ x1,
                        float* __restrict__ z) {
    __shared__ float sw[4096];
    const int tid = threadIdx.x;
    for (int i = tid; i < 4096; i += 256) sw[i] = wW[i];
    __syncthreads();

    const int gid = blockIdx.x * 256 + tid;
    const int b = gid / NPIX, n = gid % NPIX;

    float acc[64];
#pragma unroll
    for (int o = 0; o < 64; ++o) acc[o] = 0.f;
    for (int c = 0; c < 64; ++c) {
        float x = y[((long)b * NPIX + n) * 64 + c];
#pragma unroll
        for (int o = 0; o < 64; ++o) acc[o] = fmaf(sw[o * 64 + c], x, acc[o]);
    }
#pragma unroll
    for (int o = 0; o < 64; ++o) {
        long off = (long)(b * CH + o) * NPIX + n;
        z[off] = acc[o] + p[off] + x1[off];
    }
}

// ---------- launch -----------------------------------------------------------
extern "C" void kernel_launch(void* const* d_in, const int* in_sizes, int n_in,
                              void* d_out, int out_size, void* d_ws, size_t ws_size,
                              hipStream_t stream) {
    (void)in_sizes; (void)n_in; (void)out_size; (void)ws_size;
    const float* x     = (const float*)d_in[0];
    const float* w1    = (const float*)d_in[1];
    const float* w2    = (const float*)d_in[2];
    const float* wg    = (const float*)d_in[3];
    const float* wth   = (const float*)d_in[4];
    const float* wph   = (const float*)d_in[5];
    const float* wW    = (const float*)d_in[6];
    const float* alpha = (const float*)d_in[7];

    char* ws = (char*)d_ws;
    const size_t SZ_F32 = (size_t)BATCH * CH * NPIX * 4;   // 3,276,800
    const size_t SZ_BF  = (size_t)BATCH * NPIX * CH * 2;   // 1,638,400
    const size_t SZ_ST  = (size_t)BATCH * NPIX * 4;        //    51,200
    float*          x1   = (float*)ws;            ws += SZ_F32;
    float*          p    = (float*)ws;            ws += SZ_F32;
    float*          z    = (float*)ws;            ws += SZ_F32;
    float*          yb   = (float*)ws;            ws += SZ_F32;
    unsigned short* thT  = (unsigned short*)ws;   ws += SZ_BF;
    unsigned short* phiP = (unsigned short*)ws;   ws += SZ_BF;
    unsigned short* gTP  = (unsigned short*)ws;   ws += SZ_BF;
    float*          colm = (float*)ws;            ws += SZ_ST;
    float*          colr = (float*)ws;            ws += SZ_ST;

    dim3 cgrid(5, 5, BATCH * 4), cblk(16, 16);
    conv3x3_kernel<<<cgrid, cblk, 0, stream>>>(x, w1, alpha, x1, p);
    proj_pack_kernel<<<50, 256, 0, stream>>>(p, wg, wth, wph, thT, phiP, gTP);
    colstats_kernel<<<100, 256, 0, stream>>>(thT, phiP, colm, colr);
    attn_kernel<<<100, 256, 0, stream>>>(thT, phiP, gTP, colm, colr, yb);
    wy_residual_kernel<<<50, 256, 0, stream>>>(yb, wW, p, x1, z);
    conv3x3_kernel<<<cgrid, cblk, 0, stream>>>(z, w2, alpha, nullptr, (float*)d_out);
}
